// SelfAttention_8967891714364
// MI455X (gfx1250) — compile-verified, matrix operand order checked by reading, not run on hardware
//
#include <hip/hip_runtime.h>

typedef _Float16 half_t;
typedef __attribute__((ext_vector_type(16))) _Float16 v16h;
typedef __attribute__((ext_vector_type(8)))  _Float16 v8h;
typedef __attribute__((ext_vector_type(8)))  float    v8f;
typedef __attribute__((ext_vector_type(4)))  int      v4i_t;

#define B_DIM 4
#define S_DIM 2048
#define D_DIM 1024
#define A_DIM 1024

// Async global->LDS path (gfx1250), if the toolchain exposes the builtins.
#if defined(__has_builtin)
# if __has_builtin(__builtin_amdgcn_global_load_async_to_lds_b128) && \
     __has_builtin(__builtin_amdgcn_s_wait_asynccnt)
#  define USE_ASYNC_LDS 1
# endif
#endif
#ifndef USE_ASYNC_LDS
# define USE_ASYNC_LDS 0
#endif

#if USE_ASYNC_LDS
typedef __attribute__((address_space(1))) v4i_t* global_v4i_p;
typedef __attribute__((address_space(3))) v4i_t* lds_v4i_p;
#endif

// ---------------------------------------------------------------------------
// fp32 -> f16 conversion (grid-stride)
// ---------------------------------------------------------------------------
__global__ __launch_bounds__(256)
void f32_to_f16_kernel(const float* __restrict__ in, half_t* __restrict__ out, long n) {
    long i = (long)blockIdx.x * blockDim.x + threadIdx.x;
    long stride = (long)gridDim.x * blockDim.x;
    for (; i < n; i += stride) out[i] = (half_t)in[i];
}

// ---------------------------------------------------------------------------
// fp32 [R,C] -> f16 [C,R] transpose-convert (32x32 LDS tiles, 256 threads)
// ---------------------------------------------------------------------------
__global__ __launch_bounds__(256)
void transpose_f32_to_f16_kernel(const float* __restrict__ in, half_t* __restrict__ out,
                                 int R, int C) {
    __shared__ float tile[32][33];
    int tx = threadIdx.x & 31;
    int ty = threadIdx.x >> 5;          // 0..7
    int x0 = blockIdx.x * 32;
    int y0 = blockIdx.y * 32;
#pragma unroll
    for (int j = 0; j < 4; ++j)
        tile[ty + j * 8][tx] = in[(size_t)(y0 + ty + j * 8) * C + x0 + tx];
    __syncthreads();
#pragma unroll
    for (int j = 0; j < 4; ++j)
        out[(size_t)(x0 + ty + j * 8) * R + y0 + tx] =
            (half_t)tile[tx][ty + j * 8];
}

// ---------------------------------------------------------------------------
// NT WMMA GEMM:  C[M,N] = A[M,K] * Bt[N,K]^T
//   OUT_F32     : C stored as f32 (else f16)
//   STORE_TRANS : C stored transposed as Ct[N,M] (f16, packed b128 stores)
// Block 256 threads / 8 waves; tile 128x128, K-step 32, ping-pong LDS.
// Wave grid 4(M) x 2(N): each wave computes 32x64 = 2x4 WMMA tiles.
// Batched via gridDim.z with per-batch element strides.
// ---------------------------------------------------------------------------
template<bool OUT_F32, bool STORE_TRANS>
__global__ __launch_bounds__(256)
void gemm_nt_wmma_kernel(const half_t* __restrict__ Ag,
                         const half_t* __restrict__ Btg,
                         void* __restrict__ Cg,
                         int M, int N, int K,
                         long strideA, long strideB, long strideC)
{
    constexpr int TK  = 32;
    constexpr int LDT = TK + 8;                     // 40 halves = 80 B pitch
    __shared__ __align__(16) half_t As[2][128][LDT];
    __shared__ __align__(16) half_t Bs[2][128][LDT];

    const half_t* A  = Ag  + (long)blockIdx.z * strideA;
    const half_t* Bt = Btg + (long)blockIdx.z * strideB;

    const int t    = threadIdx.x;
    const int wave = t >> 5;
    const int lane = t & 31;
    const int r    = lane & 15;
    const int hi   = lane >> 4;
    const int wm   = wave & 3;          // 32 rows per M-wave
    const int wn   = wave >> 2;         // 64 cols per N-wave

    const int m0 = blockIdx.y * 128;
    const int n0 = blockIdx.x * 128;

    // both A and Bt are row-major with leading dim K: one staging routine
    auto stage = [&](half_t (*buf)[LDT], const half_t* src, int row0, int kb) {
#pragma unroll
        for (int pass = 0; pass < 2; ++pass) {
            int chunk = t + pass * 256;             // 128 rows x 4 chunks
            int rowi  = chunk >> 2;
            int colc  = (chunk & 3) << 3;
            const half_t* g = src + (size_t)(row0 + rowi) * K + kb + colc;
#if USE_ASYNC_LDS
            __builtin_amdgcn_global_load_async_to_lds_b128(
                (global_v4i_p)(void*)const_cast<half_t*>(g),
                (lds_v4i_p)(void*)&buf[rowi][colc],
                0, 0);
#else
            *(v8h*)&buf[rowi][colc] = *(const v8h*)g;
#endif
        }
    };

    v8f acc[2][4];
#pragma unroll
    for (int im = 0; im < 2; ++im)
#pragma unroll
        for (int in = 0; in < 4; ++in)
            acc[im][in] = (v8f){0.f,0.f,0.f,0.f,0.f,0.f,0.f,0.f};

    const int ntiles = K / TK;
    stage(As[0], A, m0, 0);
    stage(Bs[0], Bt, n0, 0);

    for (int it = 0; it < ntiles; ++it) {
#if USE_ASYNC_LDS
        __builtin_amdgcn_s_wait_asynccnt(0);    // async LDS writes visible
#endif
        __syncthreads();                        // buf[cur] ready; buf[cur^1] free
        const int cur = it & 1;

        if (it + 1 < ntiles) {
            stage(As[cur ^ 1], A, m0, (it + 1) * TK);
            stage(Bs[cur ^ 1], Bt, n0, (it + 1) * TK);
        }
        if (it + 2 < ntiles) {                  // gfx1250 global_prefetch_b8
            int rowi = t >> 2;
            int colc = (t & 3) << 3;
            __builtin_prefetch(A  + (size_t)(m0 + rowi) * K + (it + 2) * TK + colc, 0, 0);
            __builtin_prefetch(Bt + (size_t)(n0 + rowi) * K + (it + 2) * TK + colc, 0, 0);
        }

        // fragments: two contiguous 16B LDS reads each
        v16h af[2], bf[4];
#pragma unroll
        for (int im = 0; im < 2; ++im) {
            const v8h* p = (const v8h*)&As[cur][wm * 32 + im * 16 + r][hi * 8];
            af[im] = __builtin_shufflevector(p[0], p[2],
                     0,1,2,3,4,5,6,7,8,9,10,11,12,13,14,15);
        }
#pragma unroll
        for (int in = 0; in < 4; ++in) {
            const v8h* p = (const v8h*)&Bs[cur][wn * 64 + in * 16 + r][hi * 8];
            bf[in] = __builtin_shufflevector(p[0], p[2],
                     0,1,2,3,4,5,6,7,8,9,10,11,12,13,14,15);
        }

#pragma unroll
        for (int im = 0; im < 2; ++im)
#pragma unroll
            for (int in = 0; in < 4; ++in)
                acc[im][in] = __builtin_amdgcn_wmma_f32_16x16x32_f16(
                    false, af[im], false, bf[in],
                    (short)0, acc[im][in], false, false);
    }

    // store: VGPR v -> M = 8*hi + v, lane&15 -> N
    const size_t zc = (size_t)blockIdx.z * strideC;
#pragma unroll
    for (int im = 0; im < 2; ++im) {
#pragma unroll
        for (int in = 0; in < 4; ++in) {
            int mb = m0 + wm * 32 + im * 16 + hi * 8;
            int nb = n0 + wn * 64 + in * 16 + r;
            if (STORE_TRANS) {
                // Ct[N,M]: the 8 values are consecutive along M -> one b128 store
                v8h hv;
#pragma unroll
                for (int v = 0; v < 8; ++v) hv[v] = (half_t)acc[im][in][v];
                *(v8h*)((half_t*)Cg + zc + (size_t)nb * M + mb) = hv;
            } else {
#pragma unroll
                for (int v = 0; v < 8; ++v) {
                    size_t idx = zc + (size_t)(mb + v) * N + nb;
                    if (OUT_F32) ((float*)Cg)[idx]  = acc[im][in][v];
                    else         ((half_t*)Cg)[idx] = (half_t)acc[im][in][v];
                }
            }
        }
    }
}

// ---------------------------------------------------------------------------
// Row softmax over S=2048 f16 elements, scale folded in, in place.
// ---------------------------------------------------------------------------
__global__ __launch_bounds__(256)
void softmax_rows_kernel(half_t* __restrict__ data, int n, float scale) {
    __shared__ float red[256];
    const int t = threadIdx.x;
    half_t* p = data + (size_t)blockIdx.x * n;

    float v[8];
    float mx = -3.0e38f;
#pragma unroll
    for (int i = 0; i < 8; ++i) {
        v[i] = (float)p[t + i * 256];
        mx = fmaxf(mx, v[i]);
    }
    red[t] = mx; __syncthreads();
    for (int s = 128; s > 0; s >>= 1) {
        if (t < s) red[t] = fmaxf(red[t], red[t + s]);
        __syncthreads();
    }
    mx = red[0]; __syncthreads();

    float sum = 0.f;
#pragma unroll
    for (int i = 0; i < 8; ++i) {
        v[i] = __expf(scale * (v[i] - mx));
        sum += v[i];
    }
    red[t] = sum; __syncthreads();
    for (int s = 128; s > 0; s >>= 1) {
        if (t < s) red[t] += red[t + s];
        __syncthreads();
    }
    float inv = 1.0f / red[0];
#pragma unroll
    for (int i = 0; i < 8; ++i) p[t + i * 256] = (half_t)(v[i] * inv);
}

// ---------------------------------------------------------------------------
extern "C" void kernel_launch(void* const* d_in, const int* in_sizes, int n_in,
                              void* d_out, int out_size, void* d_ws, size_t ws_size,
                              hipStream_t stream) {
    (void)in_sizes; (void)n_in; (void)out_size; (void)ws_size;

    const float* H  = (const float*)d_in[0];
    const float* WQ = (const float*)d_in[1];
    const float* WK = (const float*)d_in[2];
    const float* WV = (const float*)d_in[3];
    float* Out = (float*)d_out;

    const long BSD = (long)B_DIM * S_DIM * D_DIM;   // 8,388,608
    const long DA  = (long)D_DIM * A_DIM;           // 1,048,576
    const long BSS = (long)B_DIM * S_DIM * S_DIM;   // 16,777,216

    half_t* ws  = (half_t*)d_ws;
    half_t* Hh  = ws;            ws += BSD;
    half_t* Wqt = ws;            ws += DA;   // [A,D] = W_Q^T
    half_t* Wkt = ws;            ws += DA;   // [A,D] = W_K^T
    half_t* Wvt = ws;            ws += DA;   // [D,D] = W_V^T
    half_t* Qh  = ws;            ws += BSD;  // [B,S,A]
    half_t* Kh  = ws;            ws += BSD;  // [B,S,A]
    half_t* Vt  = ws;            ws += BSD;  // [B,D,S] = V^T per batch
    half_t* Sh  = ws;            ws += BSS;  // scores / attn f16 [B,S,S]

    // 1) convert H; transpose-convert weights (so every GEMM is NT)
    f32_to_f16_kernel<<<2048, 256, 0, stream>>>(H, Hh, BSD);
    transpose_f32_to_f16_kernel<<<dim3(A_DIM / 32, D_DIM / 32), 256, 0, stream>>>(
        WQ, Wqt, D_DIM, A_DIM);
    transpose_f32_to_f16_kernel<<<dim3(A_DIM / 32, D_DIM / 32), 256, 0, stream>>>(
        WK, Wkt, D_DIM, A_DIM);
    transpose_f32_to_f16_kernel<<<dim3(D_DIM / 32, D_DIM / 32), 256, 0, stream>>>(
        WV, Wvt, D_DIM, D_DIM);

    // 2) Q,K projections: [8192,1024] x [1024,1024]^T (NT, f16 out)
    {
        dim3 grid(A_DIM / 128, (B_DIM * S_DIM) / 128, 1);
        gemm_nt_wmma_kernel<false, false><<<grid, 256, 0, stream>>>(
            Hh, Wqt, Qh, B_DIM * S_DIM, A_DIM, D_DIM, 0, 0, 0);
        gemm_nt_wmma_kernel<false, false><<<grid, 256, 0, stream>>>(
            Hh, Wkt, Kh, B_DIM * S_DIM, A_DIM, D_DIM, 0, 0, 0);
    }
    // 2b) V projection, stored transposed per batch: Vt[b] = (H[b] W_V)^T  [D,S]
    {
        dim3 grid(D_DIM / 128, S_DIM / 128, B_DIM);
        gemm_nt_wmma_kernel<false, true><<<grid, 256, 0, stream>>>(
            Hh, Wvt, Vt, S_DIM, D_DIM, D_DIM,
            (long)S_DIM * D_DIM, 0, (long)D_DIM * S_DIM);
    }

    // 3) scores per batch: S = Q K^T  [2048,2048] (NT, f16 out)
    {
        dim3 grid(S_DIM / 128, S_DIM / 128, B_DIM);
        gemm_nt_wmma_kernel<false, false><<<grid, 256, 0, stream>>>(
            Qh, Kh, Sh, S_DIM, S_DIM, A_DIM,
            (long)S_DIM * A_DIM, (long)S_DIM * A_DIM, (long)S_DIM * S_DIM);
    }

    // 4) softmax rows (scale = 1/sqrt(A)), in place
    softmax_rows_kernel<<<B_DIM * S_DIM, 256, 0, stream>>>(Sh, S_DIM, 0.03125f);

    // 5) output per batch: O = P V = P (Vt)^T  [2048,1024] (NT, f32 out)
    {
        dim3 grid(D_DIM / 128, S_DIM / 128, B_DIM);
        gemm_nt_wmma_kernel<true, false><<<grid, 256, 0, stream>>>(
            Sh, Vt, Out, S_DIM, D_DIM, S_DIM,
            (long)S_DIM * S_DIM, (long)D_DIM * S_DIM, (long)S_DIM * D_DIM);
    }
}
